// SelfAttention_3453153706662
// MI455X (gfx1250) — matrix operand order repaired and checked by reading, not compiled, not run on hardware
//
#include <hip/hip_runtime.h>
#include <hip/hip_bf16.h>

// ---------------- problem constants ----------------
#define B_DIM   4
#define C_DIM   512          // channels (= head dim, single head)
#define N_DIM   4096         // H*W tokens
#define M_DIM   (B_DIM * N_DIM)   // 16384 total rows
#define GROUPS_ 32
#define CG      (C_DIM / GROUPS_)      // 16 channels per group
#define GN_EPS  1e-6f
#define ATT_SCALE 0.044194173824159216f   // 512^-0.5 (folded into Wq/bq)

// flash-attention tiling
#define BR 64                // query rows per workgroup (4 waves x 16 rows)
#define BC 32                // key columns per inner tile (keeps ASYNCcnt depth low)
#define NT (N_DIM / BC)      // 128 key tiles
#define VLD (BC + 8)         // padded LDS stride for V^T rows (halfs, 80B: 16B-aligned, bank-spread)
#define PLD (BC + 8)         // padded LDS stride for P rows   (halfs)

typedef __attribute__((ext_vector_type(16))) _Float16 v16h;
typedef __attribute__((ext_vector_type(8)))  _Float16 v8h;
typedef __attribute__((ext_vector_type(8)))  float    v8f;

// ---------------------------------------------------------------------------
// Async global->LDS copy (16 bytes per lane), ASYNCcnt-tracked.
// Toolchain-probed: builtin exists; wants non-const int4* (AS1) and int4* (AS3)
// plus two immediate ints (offset, cpol).
// ---------------------------------------------------------------------------
typedef __attribute__((__vector_size__(16))) int v4i_t;
typedef __attribute__((address_space(1))) v4i_t AS1v4i;
typedef __attribute__((address_space(3))) v4i_t AS3v4i;

#if __has_builtin(__builtin_amdgcn_global_load_async_to_lds_b128)
#define ASYNC_CP16(GP, LP)                                                        \
  __builtin_amdgcn_global_load_async_to_lds_b128((AS1v4i*)(GP), (AS3v4i*)(LP), 0, 0)
#else
#define ASYNC_CP16(GP, LP)                                                        \
  asm volatile("global_load_async_to_lds_b128 %0, %1, off"                        \
               :: "v"((unsigned)(unsigned long long)(const void*)(LP)),           \
                  "v"((unsigned long long)(const void*)(GP))                      \
               : "memory")
#endif

#if __has_builtin(__builtin_amdgcn_s_wait_asynccnt)
#define WAIT_ASYNC(N) __builtin_amdgcn_s_wait_asynccnt(N)
#else
#define WAIT_ASYNC(N) asm volatile("s_wait_asynccnt " #N ::: "memory")
#endif

// ---------------------------------------------------------------------------
// WMMA fragment loaders (per CDNA5 ISA layouts, wave32)
//
// A 16x32 f16 (MxK): lane = 16*h + m; halfs = A[m, k0+8h .. +7] ++ A[m, k0+16+8h .. +7]
// B 32x16 f16 (KxN): lane = 16*h + n; halfs = B[k0+8h .. +7, n] ++ B[k0+16+8h .. +7, n]; we hold B^T
//   row-major ("Bt[n,k]"), so lane reads Bt[n, k0+8h .. +7] and Bt[n, k0+16+8h .. +7].
// C/D 16x16 f32:     vgpr r, lane = 16*h + n  ->  element (m = 8h + r, n)
// ---------------------------------------------------------------------------
static __device__ __forceinline__ v16h load_a_frag(const _Float16* p, int ld, int row0, int k0) {
  const int lane = threadIdx.x & 31;
  const int h = lane >> 4, m = lane & 15;
  const _Float16* r = p + (size_t)(row0 + m) * ld + k0 + 8 * h;
  v8h lo = *(const v8h*)(r);
  v8h hi = *(const v8h*)(r + 16);
  v16h out;
#pragma unroll
  for (int i = 0; i < 8; ++i) { out[i] = lo[i]; out[i + 8] = hi[i]; }
  return out;
}

static __device__ __forceinline__ v16h load_b_frag(const _Float16* bt, int ld, int col0, int k0) {
  const int lane = threadIdx.x & 31;
  const int h = lane >> 4, n = lane & 15;
  const _Float16* r = bt + (size_t)(col0 + n) * ld + k0 + 8 * h;
  v8h lo = *(const v8h*)(r);
  v8h hi = *(const v8h*)(r + 16);
  v16h out;
#pragma unroll
  for (int i = 0; i < 8; ++i) { out[i] = lo[i]; out[i + 8] = hi[i]; }
  return out;
}

static __device__ __forceinline__ v8f wmma_f16(v16h a, v16h b, v8f c) {
  return __builtin_amdgcn_wmma_f32_16x16x32_f16(false, a, false, b, (short)0, c, false, false);
}

// ---------------------------------------------------------------------------
// 0) f32 -> f16 weight conversion (optionally folds a scalar, e.g. 512^-0.5)
// ---------------------------------------------------------------------------
__global__ void f32_to_f16_kernel(const float* __restrict__ in, _Float16* __restrict__ out,
                                  int n, float scale) {
  int i = blockIdx.x * blockDim.x + threadIdx.x;
  if (i < n) out[i] = (_Float16)(in[i] * scale);
}

// ---------------------------------------------------------------------------
// 1) GroupNorm over [B,C,N] + transpose to xt[B,N,C] in f16.
// ---------------------------------------------------------------------------
__global__ void groupnorm_kernel(const float* __restrict__ x,
                                 const float* __restrict__ gw,
                                 const float* __restrict__ gb,
                                 _Float16* __restrict__ xt) {
  const int bg = blockIdx.x;
  const int b = bg / GROUPS_, g = bg % GROUPS_;
  const float* xg = x + ((size_t)b * C_DIM + g * CG) * N_DIM;
  const int tid = threadIdx.x;

  float s1 = 0.f, s2 = 0.f;
  for (int j = tid; j < CG * N_DIM; j += 256) { float v = xg[j]; s1 += v; s2 += v * v; }
#pragma unroll
  for (int off = 1; off < 32; off <<= 1) {
    s1 += __shfl_xor(s1, off, 32);
    s2 += __shfl_xor(s2, off, 32);
  }
  __shared__ float r1[8], r2[8];
  __shared__ float mu_s, ri_s;
  if ((tid & 31) == 0) { r1[tid >> 5] = s1; r2[tid >> 5] = s2; }
  __syncthreads();
  if (tid == 0) {
    float a = 0.f, q = 0.f;
#pragma unroll
    for (int i = 0; i < 8; ++i) { a += r1[i]; q += r2[i]; }
    const float inv_cnt = 1.0f / (float)(CG * N_DIM);
    float mu = a * inv_cnt;
    float var = q * inv_cnt - mu * mu;
    mu_s = mu;
    ri_s = rsqrtf(var + GN_EPS);
  }
  __syncthreads();
  const float mu = mu_s, ri = ri_s;
  for (int j = tid; j < CG * N_DIM; j += 256) {
    const int cc = j >> 12;
    const int n = j & (N_DIM - 1);
    const int c = g * CG + cc;
    const float y = (xg[j] - mu) * ri * gw[c] + gb[c];
    xt[((size_t)b * N_DIM + n) * C_DIM + c] = (_Float16)y;
  }
}

// ---------------------------------------------------------------------------
// 2) GEMM:  Y[M,Nd] = A[M,Kd] * W^T + bias*bscale
//    MODE 0: f16 row-major out (Q/K)
//    MODE 1: f16 TRANSPOSED out: Y[col*M + row]  (V -> V^T for attention)
//    MODE 2: f32 out + residual (final projection)
//    K-loop software-pipelined: fragments for step kk+32 are loaded before
//    the WMMAs of step kk issue, keeping global loads ahead of the MFU.
// ---------------------------------------------------------------------------
template <int MODE>
__global__ __launch_bounds__(256) void gemm_xwT_kernel(
    const _Float16* __restrict__ A, const _Float16* __restrict__ W,
    const float* __restrict__ bias, float bscale, _Float16* __restrict__ Yh,
    float* __restrict__ Yf, const float* __restrict__ res,
    int M, int Kd, int Nd) {
  const int wave = threadIdx.x >> 5;
  const int wm = wave & 1, wn = wave >> 1;
  const int row0 = blockIdx.y * 128 + wm * 64;
  const int col0 = blockIdx.x * 128 + wn * 32;

  v8f acc[4][2] = {};
  v16h a_cur[4], b_cur[2];
#pragma unroll
  for (int i = 0; i < 4; ++i) a_cur[i] = load_a_frag(A, Kd, row0 + i * 16, 0);
#pragma unroll
  for (int j = 0; j < 2; ++j) b_cur[j] = load_b_frag(W, Kd, col0 + j * 16, 0);

  for (int kk = 0; kk < Kd; kk += 32) {
    const int kn = (kk + 32 < Kd) ? kk + 32 : kk;   // tail reload is harmless
    v16h a_nxt[4], b_nxt[2];
#pragma unroll
    for (int i = 0; i < 4; ++i) a_nxt[i] = load_a_frag(A, Kd, row0 + i * 16, kn);
#pragma unroll
    for (int j = 0; j < 2; ++j) b_nxt[j] = load_b_frag(W, Kd, col0 + j * 16, kn);
#pragma unroll
    for (int i = 0; i < 4; ++i)
#pragma unroll
      for (int j = 0; j < 2; ++j) acc[i][j] = wmma_f16(a_cur[i], b_cur[j], acc[i][j]);
#pragma unroll
    for (int i = 0; i < 4; ++i) a_cur[i] = a_nxt[i];
#pragma unroll
    for (int j = 0; j < 2; ++j) b_cur[j] = b_nxt[j];
  }

  const int lane = threadIdx.x & 31;
  const int h = lane >> 4, n = lane & 15;
#pragma unroll
  for (int i = 0; i < 4; ++i)
#pragma unroll
    for (int j = 0; j < 2; ++j) {
      const int col = col0 + j * 16 + n;
      const float bb = bias[col] * bscale;
#pragma unroll
      for (int r = 0; r < 8; ++r) {
        const int row = row0 + i * 16 + 8 * h + r;
        const float v = acc[i][j][r] + bb;
        if (MODE == 0)      Yh[(size_t)row * Nd + col] = (_Float16)v;
        else if (MODE == 1) Yh[(size_t)col * M + row] = (_Float16)v;
        else {
          const size_t idx = (size_t)row * Nd + col;
          Yf[idx] = v + res[idx];
        }
      }
    }
}

// ---------------------------------------------------------------------------
// 3) Flash attention, d=512 single head.  Scale is pre-folded into Q.
//    128 threads (4 waves); each wave owns 16 full query rows.
//    Q lives in registers (16 A-fragments). K tiles [32x512] and V^T tiles
//    [512 x (32+8)] are DOUBLE-BUFFERED in LDS, staged with
//    global_load_async_to_lds_b128 (ASYNCcnt) overlapping DMA with WMMA.
//    Inner WMMA streams are software-pipelined (B fragment t+1 loaded before
//    the WMMAs of fragment t) so ds_load latency stays hidden at 1 wave/SIMD.
//    LDS: 2*32KB (K) + 2*40KB (V^T) + 5KB (P) = ~149 KB < 320 KB.
// ---------------------------------------------------------------------------
__global__ __launch_bounds__(128, 1) void flash_attn_kernel(
    const _Float16* __restrict__ Q, const _Float16* __restrict__ Kmat,
    const _Float16* __restrict__ VT, _Float16* __restrict__ ctx) {
  extern __shared__ _Float16 smem[];
  _Float16* Ks  = smem;                            // 2 * 32*512
  _Float16* VTs = smem + 2 * BC * C_DIM;           // 2 * 512*VLD
  _Float16* Ps  = VTs + 2 * C_DIM * VLD;           // 4 waves * 16 * PLD

  const int b  = blockIdx.y;
  const int rt = blockIdx.x;
  const int tid = threadIdx.x;
  const int wave = tid >> 5;
  const int lane = tid & 31;
  const int h = lane >> 4, n0 = lane & 15;

  // Q for this wave's 16 rows, kept in registers (16 A-fragments = 128 VGPRs)
  const _Float16* Qg = Q + ((size_t)b * N_DIM + (size_t)rt * BR + wave * 16) * C_DIM;
  v16h qf[16];
#pragma unroll
  for (int t = 0; t < 16; ++t) qf[t] = load_a_frag(Qg, C_DIM, 0, t * 32);

  v8f o[32] = {};                 // 16 rows x 512 ch f32 accumulator (256 VGPRs)
  float m_r[8], l_r[8];
#pragma unroll
  for (int r = 0; r < 8; ++r) { m_r[r] = -3.0e38f; l_r[r] = 0.f; }

  // async staging of one (K, V^T) tile pair: 32 async instrs per wave
  auto stage = [&](int buf, int kt) {
    const _Float16* Kb = Kmat + ((size_t)b * N_DIM + (size_t)kt * BC) * C_DIM;
    _Float16* Kl = Ks + buf * (BC * C_DIM);
#pragma unroll
    for (int i = 0; i < 16; ++i) {                 // K tile: contiguous 32 KB
      const int chunk = i * 128 + tid;             // 2048 chunks of 16B
      ASYNC_CP16(Kb + chunk * 8, Kl + chunk * 8);
    }
    const _Float16* Vb = VT + (size_t)b * N_DIM + (size_t)kt * BC;  // row c at + c*M
    _Float16* Vl = VTs + buf * (C_DIM * VLD);
#pragma unroll
    for (int i = 0; i < 16; ++i) {                 // V^T tile: 512 rows x 64B
      const int chunk = i * 128 + tid;             // 2048 chunks of 16B
      const int row = chunk >> 2, cc = chunk & 3;
      ASYNC_CP16(Vb + (size_t)row * M_DIM + cc * 8, Vl + row * VLD + cc * 8);
    }
  };

  stage(0, 0);
  for (int kt = 0; kt < NT; ++kt) {
    const int cur = kt & 1;
    if (kt + 1 < NT) {
      stage(cur ^ 1, kt + 1);   // prefetch next tile while current drains
      WAIT_ASYNC(32);           // current tile done (next tile's 32 ops outstanding)
    } else {
      WAIT_ASYNC(0);
    }
    __syncthreads();            // all waves' async data visible

    const _Float16* Kb = Ks + cur * (BC * C_DIM);
    const _Float16* Vb = VTs + cur * (C_DIM * VLD);

    // S = Q K^T (scale pre-folded into Q); software-pipelined B fragments
    v8f s[2] = {};
    {
      v16h bk0 = load_b_frag(Kb, C_DIM, 0, 0);
      v16h bk1 = load_b_frag(Kb, C_DIM, 16, 0);
#pragma unroll
      for (int t = 0; t < 16; ++t) {
        v16h nk0, nk1;
        if (t < 15) {
          nk0 = load_b_frag(Kb, C_DIM, 0, (t + 1) * 32);
          nk1 = load_b_frag(Kb, C_DIM, 16, (t + 1) * 32);
        }
        s[0] = wmma_f16(qf[t], bk0, s[0]);
        s[1] = wmma_f16(qf[t], bk1, s[1]);
        bk0 = nk0;
        bk1 = nk1;
      }
    }

    // online softmax (row m = 8h + r lives across the 16 lanes of half h)
    float corr[8];
#pragma unroll
    for (int r = 0; r < 8; ++r) {
      float mx = fmaxf(s[0][r], s[1][r]);
#pragma unroll
      for (int off = 1; off < 16; off <<= 1) mx = fmaxf(mx, __shfl_xor(mx, off, 32));
      const float mn = fmaxf(m_r[r], mx);
      corr[r] = __expf(m_r[r] - mn);
      m_r[r] = mn;
      float rs = 0.f;
#pragma unroll
      for (int j = 0; j < 2; ++j) {
        const float pv = __expf(s[j][r] - mn);
        s[j][r] = pv;
        rs += pv;
      }
#pragma unroll
      for (int off = 1; off < 16; off <<= 1) rs += __shfl_xor(rs, off, 32);
      l_r[r] = l_r[r] * corr[r] + rs;
    }
#pragma unroll
    for (int t = 0; t < 32; ++t)
#pragma unroll
      for (int r = 0; r < 8; ++r) o[t][r] *= corr[r];

    // P (C-layout f32) -> LDS f16, then re-read as one A fragment
    _Float16* Pw = Ps + wave * 16 * PLD;
#pragma unroll
    for (int r = 0; r < 8; ++r)
#pragma unroll
      for (int j = 0; j < 2; ++j)
        Pw[(8 * h + r) * PLD + j * 16 + n0] = (_Float16)s[j][r];
    asm volatile("s_wait_dscnt 0" ::: "memory");

    // O += P V ; software-pipelined V^T B fragments
    {
      v16h ap = load_a_frag(Pw, PLD, 0, 0);
      v16h bv = load_b_frag(Vb, VLD, 0, 0);
#pragma unroll
      for (int ct = 0; ct < 32; ++ct) {
        v16h nv;
        if (ct < 31) nv = load_b_frag(Vb, VLD, (ct + 1) * 16, 0);
        o[ct] = wmma_f16(ap, bv, o[ct]);
        bv = nv;
      }
    }
    __syncthreads();            // everyone done reading buffers before restage
  }

  // normalize and write ctx[b, rows, :] in f16
  float inv_l[8];
#pragma unroll
  for (int r = 0; r < 8; ++r) inv_l[r] = 1.0f / l_r[r];
  _Float16* Og = ctx + ((size_t)b * N_DIM + (size_t)rt * BR + wave * 16) * C_DIM;
#pragma unroll
  for (int ct = 0; ct < 32; ++ct)
#pragma unroll
    for (int r = 0; r < 8; ++r)
      Og[(size_t)(8 * h + r) * C_DIM + ct * 16 + n0] = (_Float16)(o[ct][r] * inv_l[r]);
}

// ---------------------------------------------------------------------------
extern "C" void kernel_launch(void* const* d_in, const int* in_sizes, int n_in,
                              void* d_out, int out_size, void* d_ws, size_t ws_size,
                              hipStream_t stream) {
  const float* x  = (const float*)d_in[0];
  const float* Wq = (const float*)d_in[1];
  const float* bq = (const float*)d_in[2];
  const float* Wk = (const float*)d_in[3];
  const float* bk = (const float*)d_in[4];
  const float* Wv = (const float*)d_in[5];
  const float* bv = (const float*)d_in[6];
  const float* Wp = (const float*)d_in[7];
  const float* bp = (const float*)d_in[8];
  const float* gw = (const float*)d_in[9];
  const float* gb = (const float*)d_in[10];

  const size_t WSZ = (size_t)C_DIM * C_DIM;          // 262144 elems
  const size_t BNC = (size_t)B_DIM * N_DIM * C_DIM;  // 8388608 elems
  _Float16* p    = (_Float16*)d_ws;
  _Float16* wq16 = p;
  _Float16* wk16 = wq16 + WSZ;
  _Float16* wv16 = wk16 + WSZ;
  _Float16* wp16 = wv16 + WSZ;
  _Float16* xt16 = wp16 + WSZ;
  _Float16* Q16  = xt16 + BNC;
  _Float16* K16  = Q16 + BNC;
  _Float16* VT16 = K16 + BNC;   // V stored TRANSPOSED: [C][B*N]
  _Float16* ctx16 = VT16 + BNC; // total ws use ~82 MB

  // weights -> f16 (attention scale folded into Wq)
  {
    const int n = (int)WSZ;
    dim3 g((n + 255) / 256);
    f32_to_f16_kernel<<<g, 256, 0, stream>>>(Wq, wq16, n, ATT_SCALE);
    f32_to_f16_kernel<<<g, 256, 0, stream>>>(Wk, wk16, n, 1.0f);
    f32_to_f16_kernel<<<g, 256, 0, stream>>>(Wv, wv16, n, 1.0f);
    f32_to_f16_kernel<<<g, 256, 0, stream>>>(Wp, wp16, n, 1.0f);
  }

  // groupnorm + transpose to [B,N,C] f16
  groupnorm_kernel<<<dim3(B_DIM * GROUPS_), 256, 0, stream>>>(x, gw, gb, xt16);

  // Q/K/V projections (V written transposed; bq scaled to match scaled Wq)
  const int M = M_DIM;             // 16384
  dim3 gg(C_DIM / 128, M / 128);   // (4, 128)
  gemm_xwT_kernel<0><<<gg, 256, 0, stream>>>(xt16, wq16, bq, ATT_SCALE, Q16, nullptr, nullptr, M, C_DIM, C_DIM);
  gemm_xwT_kernel<0><<<gg, 256, 0, stream>>>(xt16, wk16, bk, 1.0f, K16, nullptr, nullptr, M, C_DIM, C_DIM);
  gemm_xwT_kernel<1><<<gg, 256, 0, stream>>>(xt16, wv16, bv, 1.0f, VT16, nullptr, nullptr, M, C_DIM, C_DIM);

  // flash attention (double-buffered async K/V^T staging)
  const size_t smem_bytes =
      (size_t)(2 * BC * C_DIM + 2 * C_DIM * VLD + 4 * 16 * PLD) * sizeof(_Float16); // ~149 KB
  flash_attn_kernel<<<dim3(N_DIM / BR, B_DIM), 128, smem_bytes, stream>>>(Q16, K16, VT16, ctx16);

  // output projection + bias + residual (flat indexing matches the .view reshape)
  gemm_xwT_kernel<2><<<gg, 256, 0, stream>>>(ctx16, wp16, bp, 1.0f, nullptr, (float*)d_out, x, M, C_DIM, C_DIM);
}
